// LGPInstruction_3255585210489
// MI455X (gfx1250) — matrix operand order repaired and checked by reading, not compiled, not run on hardware
//
#include <hip/hip_runtime.h>
#include <hip/hip_bf16.h>
#include <math.h>

// Problem sizes
#define BT   16384   // B*T
#define VV   4096    // V
#define CC   128     // C
#define NOPS 8

typedef __attribute__((ext_vector_type(16))) __bf16 v16bf;
typedef __attribute__((ext_vector_type(8)))  float  v8f;

union FragBF { v16bf v; uint4 q[2]; };

__device__ inline v8f v8f_zero() {
    v8f z;
#pragma unroll
    for (int i = 0; i < 8; ++i) z[i] = 0.0f;
    return z;
}

__device__ inline unsigned short f2bfbits(float f) {
    unsigned u = __float_as_uint(f);
    u += 0x7FFFu + ((u >> 16) & 1u);   // round-to-nearest-even
    return (unsigned short)(u >> 16);
}
__device__ inline unsigned bfpack(float a, float b) {
    return (unsigned)f2bfbits(a) | ((unsigned)f2bfbits(b) << 16);
}

__device__ inline v8f wmma_bf16(v16bf a, v16bf b, v8f c) {
    return __builtin_amdgcn_wmma_f32_16x16x32_bf16(false, a, false, b, (short)0, c,
                                                   false, false);
}

// Async global->LDS 16B copy per lane (CDNA5). VDST operand = LDS byte address
// (low 32 bits of the generic shared pointer per the aperture mapping).
__device__ inline void async_ld_b128(unsigned lds_off, const void* gaddr) {
    asm volatile("global_load_async_to_lds_b128 %0, %1, off"
                 :: "v"(lds_off), "v"(gaddr)
                 : "memory");
}
__device__ inline void wait_async0() {
#if __has_builtin(__builtin_amdgcn_s_wait_asynccnt)
    __builtin_amdgcn_s_wait_asynccnt(0);
#else
    asm volatile("s_wait_asynccnt 0x0" ::: "memory");
#endif
}

// A fragment: 16x32 bf16 row-major tile in LDS (stride in bf16 elems, 16B-aligned rows).
// lane l: m = l&15, kb = (l>>4)*8 ; elems j<8 -> K=kb+j ; j>=8 -> K=16+kb+(j-8)
__device__ inline v16bf frag_a(const __bf16* base, int stride) {
    int lane = threadIdx.x & 31;
    int m    = lane & 15;
    int kb   = (lane >> 4) << 3;
    const __bf16* r = base + m * stride;
    FragBF f;
    f.q[0] = *(const uint4*)(r + kb);
    f.q[1] = *(const uint4*)(r + 16 + kb);
    return f.v;
}
// B fragment: 32x16 bf16 tile stored COLUMN-major in LDS as [n][k] (stride elems).
// lane l: n = l&15, kb = (l>>4)*8 ; elems j<8 -> K=kb+j ; j>=8 -> K=16+kb+(j-8)
__device__ inline v16bf frag_b(const __bf16* base, int stride) {
    int lane = threadIdx.x & 31;
    int n    = lane & 15;
    int kb   = (lane >> 4) << 3;
    const __bf16* r = base + n * stride;
    FragBF f;
    f.q[0] = *(const uint4*)(r + kb);
    f.q[1] = *(const uint4*)(r + 16 + kb);
    return f.v;
}

__device__ inline float apply_nl(int i, float h) {
    switch (i) {
        case 0: return h;
        case 1: return fmaxf(h, 0.0f);
        case 2: return 0.5f * h * (1.0f + erff(h * 0.70710678118654752f));
        case 3: return h * h;
        case 4: return -h;
        case 5: return fabsf(h);
        case 6: return tanhf(h);
        case 7: return 1.0f / (1.0f + __expf(-h));
    }
    return h;
}

// ---------------- prep kernels ----------------

// write_w[v][c] = out_scale * dot(basis[v,:], write_coeffs[c,:])  -> bf16
__global__ __launch_bounds__(256) void prep_ww_kernel(
    const float* __restrict__ basis, const float* __restrict__ wc,
    const float* __restrict__ out_scale, unsigned short* __restrict__ ww) {
    int idx = blockIdx.x * 256 + threadIdx.x;   // V*C = 524288 threads
    int v = idx >> 7, c = idx & 127;
    const float* b = basis + (size_t)v * CC;
    const float* w = wc + (size_t)c * CC;
    float s = 0.0f;
#pragma unroll 8
    for (int k = 0; k < CC; ++k) s = fmaf(b[k], w[k], s);
    ww[idx] = f2bfbits(s * out_scale[0]);
}

__global__ __launch_bounds__(256) void conv_opw_kernel(
    const float* __restrict__ opw, unsigned short* __restrict__ o) {
    int idx = blockIdx.x * 256 + threadIdx.x;   // 8*128*128 = 131072
    o[idx] = f2bfbits(opw[idx]);
}

// read_w = softmax(basis @ read_coeffs.T, axis=0): one block per column c.
__global__ __launch_bounds__(256) void read_softmax_kernel(
    const float* __restrict__ basis, const float* __restrict__ rc,
    unsigned short* __restrict__ rw) {
    int c   = blockIdx.x;
    int tid = threadIdx.x;
    __shared__ float coeff[CC];
    __shared__ float red[256];
    if (tid < CC) coeff[tid] = rc[(size_t)c * CC + tid];
    __syncthreads();

    float logit[16];
    float mx = -3.4e38f;
#pragma unroll
    for (int j = 0; j < 16; ++j) {
        const float* b = basis + (size_t)(tid + j * 256) * CC;
        float s = 0.0f;
#pragma unroll 8
        for (int k = 0; k < CC; ++k) s = fmaf(b[k], coeff[k], s);
        logit[j] = s;
        mx = fmaxf(mx, s);
    }
    red[tid] = mx; __syncthreads();
    for (int st = 128; st > 0; st >>= 1) {
        if (tid < st) red[tid] = fmaxf(red[tid], red[tid + st]);
        __syncthreads();
    }
    mx = red[0]; __syncthreads();

    float sum = 0.0f;
#pragma unroll
    for (int j = 0; j < 16; ++j) { logit[j] = __expf(logit[j] - mx); sum += logit[j]; }
    red[tid] = sum; __syncthreads();
    for (int st = 128; st > 0; st >>= 1) {
        if (tid < st) red[tid] += red[tid + st];
        __syncthreads();
    }
    sum = red[0];
    float inv = 1.0f / sum;
#pragma unroll
    for (int j = 0; j < 16; ++j)
        rw[(size_t)(tid + j * 256) * CC + c] = f2bfbits(logit[j] * inv);
}

// ---------------- fused READ gemm + OP BANK ----------------
// 64-row M tile per block, 8 waves: wave w -> m-subtile (w&3), n-half (w>>2).
#define MT 64
__global__ __launch_bounds__(256) void fused_read_opbank(
    const float* __restrict__ x, const unsigned short* __restrict__ rw,
    const unsigned short* __restrict__ opw, const float* __restrict__ op_logits,
    const float* __restrict__ op_biases, unsigned short* __restrict__ vals_out) {
    __shared__ __align__(16) unsigned char smem[MT * 136 * 2 + MT * 40 * 2 + CC * 40 * 2];
    __bf16* valsT = (__bf16*)smem;                                  // 64 x 136
    __bf16* At    = (__bf16*)(smem + MT * 136 * 2);                 // 64 x 40
    __bf16* Bt    = (__bf16*)(smem + MT * 136 * 2 + MT * 40 * 2);   // 128 x 40
    unsigned short* BtS = (unsigned short*)Bt;

    int tid  = threadIdx.x;
    int wave = tid >> 5, lane = tid & 31;
    int mrow = (wave & 3) * 16;
    int ntb  = (wave >> 2) * 4;
    int mblock = blockIdx.x * MT;

    v8f acc[4];
#pragma unroll
    for (int i = 0; i < 4; ++i) acc[i] = v8f_zero();

    const unsigned* rwU = (const unsigned*)rw;
    for (int kk = 0; kk < VV; kk += 32) {
        // A tile: x[mblock..+63][kk..+31] f32 -> bf16 (512 float4, 2 per thread)
#pragma unroll
        for (int i = 0; i < 2; ++i) {
            int e = tid + i * 256;
            int row = e >> 3, c4 = e & 7;
            float4 v = ((const float4*)(x + (size_t)(mblock + row) * VV + kk))[c4];
            uint2 p; p.x = bfpack(v.x, v.y); p.y = bfpack(v.z, v.w);
            *(uint2*)(At + row * 40 + c4 * 4) = p;
        }
        // prefetch next K tile of x while this one computes
        if (kk + 32 < VV) {
#pragma unroll
            for (int i = 0; i < 2; ++i) {
                int e = tid + i * 256;
                int row = e >> 3, c4 = e & 7;
                __builtin_prefetch(x + (size_t)(mblock + row) * VV + kk + 32 + c4 * 4, 0, 1);
            }
        }
        // B tile (col-major [n][k]): read_w rows kk..kk+31, all 128 cols
#pragma unroll
        for (int i = 0; i < 8; ++i) {
            int e = tid + i * 256;          // 2048 uints
            int k = e >> 6, cu = e & 63;
            unsigned w = rwU[(size_t)(kk + k) * 64 + cu];
            BtS[(2 * cu) * 40 + k]     = (unsigned short)(w & 0xFFFFu);
            BtS[(2 * cu + 1) * 40 + k] = (unsigned short)(w >> 16);
        }
        __syncthreads();
        v16bf af = frag_a(At + mrow * 40, 40);
#pragma unroll
        for (int nt = 0; nt < 4; ++nt) {
            v16bf bf = frag_b(Bt + (ntb + nt) * 16 * 40, 40);
            acc[nt] = wmma_bf16(af, bf, acc[nt]);
        }
        __syncthreads();
    }

    // stage values tile (bf16) for op-bank GEMMs
    unsigned short* valsTS = (unsigned short*)valsT;
#pragma unroll
    for (int nt = 0; nt < 4; ++nt) {
        int n = (ntb + nt) * 16 + (lane & 15);
#pragma unroll
        for (int r = 0; r < 8; ++r) {
            int m = mrow + r + ((lane >> 4) << 3);
            valsTS[m * 136 + n] = f2bfbits(acc[nt][r]);
        }
    }

    // softmax over 8 op logits (per thread, broadcast scalar loads)
    float wsm[NOPS];
    {
        float mx = op_logits[0];
#pragma unroll
        for (int i = 1; i < NOPS; ++i) mx = fmaxf(mx, op_logits[i]);
        float s = 0.0f;
#pragma unroll
        for (int i = 0; i < NOPS; ++i) { wsm[i] = __expf(op_logits[i] - mx); s += wsm[i]; }
        float inv = 1.0f / s;
#pragma unroll
        for (int i = 0; i < NOPS; ++i) wsm[i] *= inv;
    }

    v8f res[4];
#pragma unroll
    for (int i = 0; i < 4; ++i) res[i] = v8f_zero();
    __syncthreads();

    const unsigned* opwU = (const unsigned*)opw;
#pragma unroll
    for (int op = 0; op < NOPS; ++op) {
        v8f h[4];
#pragma unroll
        for (int i = 0; i < 4; ++i) h[i] = v8f_zero();
        for (int ks = 0; ks < 4; ++ks) {
            // B tile: opw[op][(ks*32+k)][n] -> Bt[n][k]
#pragma unroll
            for (int i = 0; i < 8; ++i) {
                int e = tid + i * 256;
                int k = e >> 6, nu = e & 63;
                unsigned w = opwU[(size_t)op * 8192 + (size_t)(ks * 32 + k) * 64 + nu];
                BtS[(2 * nu) * 40 + k]     = (unsigned short)(w & 0xFFFFu);
                BtS[(2 * nu + 1) * 40 + k] = (unsigned short)(w >> 16);
            }
            __syncthreads();
            v16bf af = frag_a(valsT + mrow * 136 + ks * 32, 136);
#pragma unroll
            for (int nt = 0; nt < 4; ++nt) {
                v16bf bf = frag_b(Bt + (ntb + nt) * 16 * 40, 40);
                h[nt] = wmma_bf16(af, bf, h[nt]);
            }
            __syncthreads();
        }
        float wo = wsm[op];
#pragma unroll
        for (int nt = 0; nt < 4; ++nt) {
            int n = (ntb + nt) * 16 + (lane & 15);
            float bias = op_biases[op * CC + n];
#pragma unroll
            for (int r = 0; r < 8; ++r) {
                float v = h[nt][r] + bias;
                res[nt][r] += wo * apply_nl(op, v);
            }
        }
    }

    // values2 -> workspace (bf16, row-major BTxC)
#pragma unroll
    for (int nt = 0; nt < 4; ++nt) {
        int n = (ntb + nt) * 16 + (lane & 15);
#pragma unroll
        for (int r = 0; r < 8; ++r) {
            int m = mrow + r + ((lane >> 4) << 3);
            vals_out[(size_t)(mblock + m) * CC + n] = f2bfbits(res[nt][r]);
        }
    }
}

// ---------------- WRITE gemm: out = values2 @ write_w^T (scale pre-folded) ----------------
// LDS staging done with GLOBAL_LOAD_ASYNC_TO_LDS_B128 (ASYNCcnt path).
__global__ __launch_bounds__(256) void write_gemm(
    const unsigned short* __restrict__ vals, const unsigned short* __restrict__ ww,
    float* __restrict__ out) {
    __shared__ __align__(16) unsigned char smem[128 * 136 * 2 + 128 * 40 * 2];
    __bf16* valsT = (__bf16*)smem;                       // 128 x 136
    __bf16* Bt    = (__bf16*)(smem + 128 * 136 * 2);     // 128 x 40
    unsigned short* valsTS = (unsigned short*)valsT;
    unsigned short* BtS    = (unsigned short*)Bt;

    int tid = threadIdx.x, wave = tid >> 5, lane = tid & 31;
    int mblock = blockIdx.y * 128;
    int vblock = blockIdx.x * 128;

    // stage values2 tile (128 x 128 bf16 = 2048 x 16B chunks) via async DMA
#pragma unroll
    for (int i = 0; i < 8; ++i) {
        int e = tid + i * 256;
        int row = e >> 4, q = e & 15;
        async_ld_b128((unsigned)(uintptr_t)(valsTS + row * 136 + q * 8),
                      (const unsigned char*)vals + (size_t)(mblock + row) * 256 + q * 16);
    }

    v8f acc[8];
#pragma unroll
    for (int i = 0; i < 8; ++i) acc[i] = v8f_zero();

    for (int ks = 0; ks < 4; ++ks) {
        // B[k][n] = write_w[vblock+n][ks*32+k]; per n: 64B contiguous -> 4 x 16B chunks
#pragma unroll
        for (int i = 0; i < 2; ++i) {
            int e = tid + i * 256;           // 512 chunks
            int n = e >> 2, q = e & 3;
            async_ld_b128((unsigned)(uintptr_t)(BtS + n * 40 + q * 8),
                          (const unsigned char*)ww + (size_t)(vblock + n) * 256 +
                              (size_t)ks * 64 + q * 16);
        }
        wait_async0();
        __syncthreads();
        v16bf af = frag_a(valsT + wave * 16 * 136 + ks * 32, 136);
#pragma unroll
        for (int nt = 0; nt < 8; ++nt) {
            v16bf bf = frag_b(Bt + nt * 16 * 40, 40);
            acc[nt] = wmma_bf16(af, bf, acc[nt]);
        }
        __syncthreads();
    }

#pragma unroll
    for (int nt = 0; nt < 8; ++nt) {
        int n = vblock + nt * 16 + (lane & 15);
#pragma unroll
        for (int r = 0; r < 8; ++r) {
            int m = mblock + wave * 16 + r + ((lane >> 4) << 3);
            out[(size_t)m * VV + n] = acc[nt][r];
        }
    }
}

extern "C" void kernel_launch(void* const* d_in, const int* in_sizes, int n_in,
                              void* d_out, int out_size, void* d_ws, size_t ws_size,
                              hipStream_t stream) {
    (void)in_sizes; (void)n_in; (void)out_size; (void)ws_size;
    const float* x     = (const float*)d_in[0];
    const float* basis = (const float*)d_in[1];
    const float* rc    = (const float*)d_in[2];
    const float* wc    = (const float*)d_in[3];
    const float* opl   = (const float*)d_in[4];
    const float* opw   = (const float*)d_in[5];
    const float* opb   = (const float*)d_in[6];
    const float* osc   = (const float*)d_in[7];
    float* out = (float*)d_out;

    unsigned char* ws = (unsigned char*)d_ws;
    unsigned short* rw   = (unsigned short*)(ws);                              // 1 MB
    unsigned short* ww   = (unsigned short*)(ws + (1u << 20));                 // 1 MB
    unsigned short* opwb = (unsigned short*)(ws + (2u << 20));                 // 256 KB
    unsigned short* vals = (unsigned short*)(ws + (2u << 20) + (1u << 19));    // 4 MB

    prep_ww_kernel<<<(VV * CC) / 256, 256, 0, stream>>>(basis, wc, osc, ww);
    conv_opw_kernel<<<(NOPS * CC * CC) / 256, 256, 0, stream>>>(opw, opwb);
    read_softmax_kernel<<<CC, 256, 0, stream>>>(basis, rc, rw);
    fused_read_opbank<<<BT / MT, 256, 0, stream>>>(x, rw, opwb, opl, opb, vals);
    write_gemm<<<dim3(VV / 128, BT / 128), 256, 0, stream>>>(vals, ww, out);
}
